// DiffusionTransformer_77043123355884
// MI455X (gfx1250) — matrix operand order repaired and checked by reading, not compiled, not run on hardware
//
#include <hip/hip_runtime.h>
#include <math.h>

// ---------------- types ----------------
typedef __bf16 bf16_t;
typedef __bf16 v16bf __attribute__((ext_vector_type(16)));
typedef __bf16 v8bf  __attribute__((ext_vector_type(8)));
typedef float  v8f   __attribute__((ext_vector_type(8)));

#define SHUF16(lo, hi) __builtin_shufflevector(lo, hi, 0,1,2,3,4,5,6,7,8,9,10,11,12,13,14,15)

// ---------------- problem constants ----------------
static constexpr int N   = 1536;
static constexpr int C   = 384;
static constexpr int CP  = 16;
static constexpr int H   = 16;
static constexpr int DH  = 24;
static constexpr int DHP = 32;   // DH padded to 32 for WMMA K
static constexpr int L   = 4;
static constexpr int HID = 768;
static constexpr int CC  = C * C;
static constexpr int CH  = C * HID;
static constexpr int NC  = N * C;
static constexpr int NHID = N * HID;

__device__ __forceinline__ float sigm(float x) { return 1.0f / (1.0f + __expf(-x)); }

// ---- DPP16 cross-lane reductions within each 16-lane group (no LDS, pure VALU) ----
// ctrl: 0xB1 = quad_perm[1,0,3,2] (xor1), 0x4E = quad_perm[2,3,0,1] (xor2),
//       0x141 = row_half_mirror (xor7), 0x140 = row_mirror (xor15)
template <int CTRL>
__device__ __forceinline__ float dpp16_f(float x) {
    return __int_as_float(__builtin_amdgcn_update_dpp(
        __float_as_int(x), __float_as_int(x), CTRL, 0xf, 0xf, false));
}
__device__ __forceinline__ float red16_max(float x) {
    x = fmaxf(x, dpp16_f<0xB1>(x));
    x = fmaxf(x, dpp16_f<0x4E>(x));
    x = fmaxf(x, dpp16_f<0x141>(x));
    x = fmaxf(x, dpp16_f<0x140>(x));
    return x;
}
__device__ __forceinline__ float red16_sum(float x) {
    x += dpp16_f<0xB1>(x);
    x += dpp16_f<0x4E>(x);
    x += dpp16_f<0x141>(x);
    x += dpp16_f<0x140>(x);
    return x;
}

// ================= elementwise / prep kernels =================

__global__ void cvt_bf_kernel(const float* __restrict__ s, bf16_t* __restrict__ d, int n) {
    int i = blockIdx.x * 256 + threadIdx.x;
    if (i < n) d[i] = (bf16_t)s[i];
}

__global__ void zero_bf_kernel(bf16_t* __restrict__ p, int n) {
    int i = blockIdx.x * 256 + threadIdx.x;
    if (i < n) p[i] = (bf16_t)0.0f;
}

// out = sigmoid(gpre) * aln + add   (bf16 out)
__global__ void ew_combine_kernel(const float* __restrict__ gpre, const float* __restrict__ aln,
                                  const float* __restrict__ add, bf16_t* __restrict__ out, int n) {
    int i = blockIdx.x * 256 + threadIdx.x;
    if (i < n) out[i] = (bf16_t)(sigm(gpre[i]) * aln[i] + add[i]);
}

// LayerNorm per row (no affine), one wave32 per row; optional f32 / bf16 outputs
__global__ __launch_bounds__(32) void ln_rows_kernel(const float* __restrict__ src,
                                                     float* __restrict__ outf,
                                                     bf16_t* __restrict__ outb, int cols) {
    int row = blockIdx.x;
    int lane = threadIdx.x;
    const float* r = src + (size_t)row * cols;
    float s = 0.f, sq = 0.f;
    for (int c = lane; c < cols; c += 32) { float v = r[c]; s += v; sq += v * v; }
    for (int off = 16; off > 0; off >>= 1) { s += __shfl_xor(s, off); sq += __shfl_xor(sq, off); }
    float mean = s / cols;
    float var  = sq / cols - mean * mean;
    float rstd = rsqrtf(var + 1e-5f);
    for (int c = lane; c < cols; c += 32) {
        float y = (r[c] - mean) * rstd;
        if (outf) outf[(size_t)row * cols + c] = y;
        if (outb) outb[(size_t)row * cols + c] = (bf16_t)y;
    }
}

// weight prep: dst[j*K + c] = bf16(src[c*Nc + j] * rowscale[c])   (transpose + optional fold)
__global__ void prep_w_kernel(const float* __restrict__ src, const float* __restrict__ rowscale,
                              bf16_t* __restrict__ dst, int K, int Nc) {
    int idx = blockIdx.x * 256 + threadIdx.x;
    if (idx >= K * Nc) return;
    int c = idx / Nc, j = idx - c * Nc;
    float v = src[idx];
    if (rowscale) v *= rowscale[c];
    dst[(size_t)j * K + c] = (bf16_t)v;
}

// zconst[l][h] = sum_c at_zln_b[l][c] * at_Wb[l][c][h]
__global__ void zconst_kernel(const float* __restrict__ zlnb, const float* __restrict__ wb,
                              float* __restrict__ out) {
    int i = threadIdx.x;
    if (i >= L * H) return;
    int l = i / H, h = i - l * H;
    float acc = 0.f;
    for (int c = 0; c < CP; c++) acc += zlnb[l * CP + c] * wb[(size_t)(l * CP + c) * H + h];
    out[i] = acc;
}

// zb[h][q][k] = LN(z[q,k,:]) @ (g*Wb)[:,h] + zconst[h] + (mask[k]-1)*1e9
__global__ __launch_bounds__(256) void zbias_kernel(const float* __restrict__ z,
                                                    const bf16_t* __restrict__ wbp, // [H][CP]
                                                    const float* __restrict__ zc,   // [H]
                                                    const float* __restrict__ mask,
                                                    float* __restrict__ zb) {
    __shared__ float w[H][CP];
    __shared__ float zcs[H];
    int t = threadIdx.x;
    if (t < H * CP) w[t / CP][t % CP] = (float)wbp[t];
    if (t < H) zcs[t] = zc[t];
    __syncthreads();
    size_t idx = (size_t)blockIdx.x * 256 + t;   // < N*N
    int q = (int)(idx / N);
    int k = (int)(idx - (size_t)q * N);
    const float* zp = z + idx * CP;
    float v[CP];
    float s = 0.f;
#pragma unroll
    for (int c = 0; c < CP; c++) { v[c] = zp[c]; s += v[c]; }
    float mean = s * (1.0f / CP);
    float sq = 0.f;
#pragma unroll
    for (int c = 0; c < CP; c++) { float d = v[c] - mean; sq += d * d; }
    float rstd = rsqrtf(sq * (1.0f / CP) + 1e-5f);
#pragma unroll
    for (int c = 0; c < CP; c++) v[c] = (v[c] - mean) * rstd;
    float mb = (mask[k] - 1.0f) * 1e9f;
#pragma unroll
    for (int h = 0; h < H; h++) {
        float acc = zcs[h];
#pragma unroll
        for (int c = 0; c < CP; c++) acc += v[c] * w[h][c];
        zb[(size_t)h * N * N + idx] = acc + mb;
    }
}

// ================= bf16 WMMA GEMM =================
// A: [M][K] row-major bf16.  W: [Nc][K] row-major bf16 (pre-transposed weights).
// Each wave computes a 16x64 output tile via 4x v_wmma_f32_16x16x32_bf16, K-loop step 32.
// Epilogues: 0 plain f32 | 1 sigmoid f32 | 3 Q/K head layout [H][M][32] bf16
//            4 V transposed [H][32][M] bf16 | 5 gated dual (f32 + bf16)
//            6 SwiGLU bf16 | 7 residual f32
template <int EPI>
__global__ __launch_bounds__(128) void gemm_k(const bf16_t* __restrict__ A,
                                              const bf16_t* __restrict__ W,
                                              const float* __restrict__ bias, float alpha,
                                              int M, int K, int Nc,
                                              float* __restrict__ outf, bf16_t* __restrict__ outb,
                                              const float* __restrict__ aux0,
                                              const float* __restrict__ aux1) {
    const int lane = threadIdx.x & 31;
    const int wave = threadIdx.x >> 5;
    const int tiles_n = Nc >> 6;
    const int tile = blockIdx.x * 4 + wave;
    const int rt = tile / tiles_n;
    const int ct = tile - rt * tiles_n;
    const int row0 = rt << 4;
    const int col0 = ct << 6;
    const int l15 = lane & 15;
    const int hiA = (lane >> 4) << 3;   // +8  K-offset for A chunks
    const int hiB = (lane >> 4) << 4;   // +16 K-offset (unused; B loads use hiA like A)

    const bf16_t* Ap = A + (size_t)(row0 + l15) * K + hiA;

    v8f acc[4];
#pragma unroll
    for (int t = 0; t < 4; t++) acc[t] = (v8f){0.f, 0.f, 0.f, 0.f, 0.f, 0.f, 0.f, 0.f};

    for (int k0 = 0; k0 < K; k0 += 32) {
        v8bf alo = *(const v8bf*)(Ap + k0);
        v8bf ahi = *(const v8bf*)(Ap + k0 + 16);
        v16bf a = SHUF16(alo, ahi);
#pragma unroll
        for (int t = 0; t < 4; t++) {
            int col = col0 + (t << 4) + l15;
            v16bf b = SHUF16(*(const v8bf*)(W + (size_t)col * K + k0 + hiA), *(const v8bf*)(W + (size_t)col * K + k0 + hiA + 16));
            acc[t] = __builtin_amdgcn_wmma_f32_16x16x32_bf16(false, a, false, b, (short)0,
                                                             acc[t], false, false);
        }
    }

    const int rbase = row0 + ((lane >> 4) << 3);
#pragma unroll
    for (int t = 0; t < 4; t++) {
        int col = col0 + (t << 4) + l15;
        float bv = bias ? bias[col] : 0.0f;
#pragma unroll
        for (int r = 0; r < 8; r++) {
            int row = rbase + r;
            float v = alpha * (acc[t][r] + bv);
            size_t idx = (size_t)row * Nc + col;
            if constexpr (EPI == 0) {
                outf[idx] = v;
            } else if constexpr (EPI == 1) {
                outf[idx] = sigm(v);
            } else if constexpr (EPI == 3) {
                int h = col / DH, d = col - h * DH;
                outb[((size_t)h * M + row) * DHP + d] = (bf16_t)v;
            } else if constexpr (EPI == 4) {
                int h = col / DH, d = col - h * DH;
                outb[(size_t)(h * DHP + d) * M + row] = (bf16_t)v;
            } else if constexpr (EPI == 5) {
                float v2 = aux0[idx] * v;
                outf[idx] = v2;
                outb[idx] = (bf16_t)v2;
            } else if constexpr (EPI == 6) {
                float hv = aux0[idx];
                outb[idx] = (bf16_t)(hv * sigm(hv) * v);
            } else if constexpr (EPI == 7) {
                outf[idx] = aux0[idx] + aux1[idx] * v;
            }
        }
    }
}

// ================= flash attention (one wave per (head, 16-q tile)) =================
// qb/kb: [H][N][32] bf16 (d padded w/ zeros), vt: [H][32][N] bf16, zb: [H][N][N] f32 (incl mask),
// gateh: [N][C] f32 (pre-sigmoid), og: [N][C] bf16 output = sigmoid(gate)*O.
__global__ __launch_bounds__(128) void attn_kernel(const bf16_t* __restrict__ qb,
                                                   const bf16_t* __restrict__ kb,
                                                   const bf16_t* __restrict__ vt,
                                                   const float* __restrict__ zb,
                                                   const float* __restrict__ gateh,
                                                   bf16_t* __restrict__ og) {
    __shared__ __align__(64) bf16_t pb[4][16][32];
    const int lane = threadIdx.x & 31;
    const int wave = threadIdx.x >> 5;
    const int task = blockIdx.x * 4 + wave;
    const int qtiles = N >> 4;
    const int h  = task / qtiles;
    const int qt = task - h * qtiles;
    const int q0 = qt << 4;
    const int l15 = lane & 15;
    const int hi8  = (lane >> 4) << 3;
    const int hi16 = (lane >> 4) << 4;

    // Q A-fragment (16 rows x 32 padded-d), loaded once
    const bf16_t* qp = qb + ((size_t)h * N + q0 + l15) * DHP + hi16;
    v16bf qa = SHUF16(*(const v8bf*)qp, *(const v8bf*)(qp + 8));

    float mrow[8], lrow[8];
#pragma unroll
    for (int r = 0; r < 8; r++) { mrow[r] = -1e30f; lrow[r] = 0.f; }
    v8f o0 = (v8f){0.f,0.f,0.f,0.f,0.f,0.f,0.f,0.f};
    v8f o1 = (v8f){0.f,0.f,0.f,0.f,0.f,0.f,0.f,0.f};

    const float* zr = zb + (size_t)h * N * N;
    const bf16_t* vbase = vt + (size_t)h * DHP * N;

    for (int kb0 = 0; kb0 < N; kb0 += 32) {
        const bf16_t* kp = kb + ((size_t)h * N + kb0 + l15) * DHP + hi16;
        if (kb0 + 32 < N) __builtin_prefetch((const void*)(kp + 32 * DHP), 0, 0);
        v16bf B0 = *(const v16bf*)kp;
        v16bf B1 = *(const v16bf*)(kp + 16 * DHP);

        v8f c0, c1;
#pragma unroll
        for (int r = 0; r < 8; r++) {
            size_t qr = (size_t)(q0 + r + hi8) * N + kb0 + l15;
            c0[r] = zr[qr];
            c1[r] = zr[qr + 16];
        }
        v8f s0 = __builtin_amdgcn_wmma_f32_16x16x32_bf16(false, qa, false, B0, (short)0, c0, false, false);
        v8f s1 = __builtin_amdgcn_wmma_f32_16x16x32_bf16(false, qa, false, B1, (short)0, c1, false, false);

#pragma unroll
        for (int r = 0; r < 8; r++) {
            float mx = red16_max(fmaxf(s0[r], s1[r]));          // DPP16 butterfly (xor1/2/7/15)
            float mn = fmaxf(mrow[r], mx);
            float corr = __expf(mrow[r] - mn);
            float p0 = __expf(s0[r] - mn);
            float p1 = __expf(s1[r] - mn);
            float rs = red16_sum(p0 + p1);                      // DPP16 butterfly
            lrow[r] = lrow[r] * corr + rs;
            mrow[r] = mn;
            o0[r] = o0[r] * corr;
            o1[r] = o1[r] * corr;
            pb[wave][r + hi8][l15]      = (bf16_t)p0;
            pb[wave][r + hi8][l15 + 16] = (bf16_t)p1;
        }
        asm volatile("s_wait_dscnt 0" ::: "memory");  // wave-internal LDS RAW (C-layout -> A-layout)

        v16bf pa = SHUF16(*(const v8bf*)&pb[wave][l15][hi16],
                          *(const v8bf*)&pb[wave][l15][hi16 + 8]);
        const bf16_t* vp = vbase + (size_t)l15 * N + kb0 + hi16;
        v16bf V0 = *(const v16bf*)vp;
        v16bf V1 = *(const v16bf*)(vp + (size_t)16 * N);
        o0 = __builtin_amdgcn_wmma_f32_16x16x32_bf16(false, pa, false, V0, (short)0, o0, false, false);
        o1 = __builtin_amdgcn_wmma_f32_16x16x32_bf16(false, pa, false, V1, (short)0, o1, false, false);
    }

#pragma unroll
    for (int r = 0; r < 8; r++) {
        int row = q0 + r + hi8;
        float invl = 1.0f / lrow[r];
        {
            int col = h * DH + l15;                 // d in [0,16)
            float g = sigm(gateh[(size_t)row * C + col]);
            og[(size_t)row * C + col] = (bf16_t)(o0[r] * invl * g);
        }
        if (l15 < 8) {
            int col = h * DH + 16 + l15;            // d in [16,24)
            float g = sigm(gateh[(size_t)row * C + col]);
            og[(size_t)row * C + col] = (bf16_t)(o1[r] * invl * g);
        }
    }
}

// ================= host side =================

static void launch_gemm(int epi, const bf16_t* A, const bf16_t* W, const float* bias, float alpha,
                        int M, int K, int Nc, float* outf, bf16_t* outb,
                        const float* aux0, const float* aux1, hipStream_t st) {
    int tiles = (M / 16) * (Nc / 64);
    dim3 grid(tiles / 4), blk(128);
    switch (epi) {
        case 0: gemm_k<0><<<grid, blk, 0, st>>>(A, W, bias, alpha, M, K, Nc, outf, outb, aux0, aux1); break;
        case 1: gemm_k<1><<<grid, blk, 0, st>>>(A, W, bias, alpha, M, K, Nc, outf, outb, aux0, aux1); break;
        case 3: gemm_k<3><<<grid, blk, 0, st>>>(A, W, bias, alpha, M, K, Nc, outf, outb, aux0, aux1); break;
        case 4: gemm_k<4><<<grid, blk, 0, st>>>(A, W, bias, alpha, M, K, Nc, outf, outb, aux0, aux1); break;
        case 5: gemm_k<5><<<grid, blk, 0, st>>>(A, W, bias, alpha, M, K, Nc, outf, outb, aux0, aux1); break;
        case 6: gemm_k<6><<<grid, blk, 0, st>>>(A, W, bias, alpha, M, K, Nc, outf, outb, aux0, aux1); break;
        case 7: gemm_k<7><<<grid, blk, 0, st>>>(A, W, bias, alpha, M, K, Nc, outf, outb, aux0, aux1); break;
    }
}

static inline void* wsalloc(char*& p, size_t bytes) {
    void* r = (void*)p;
    p += (bytes + 255) & ~(size_t)255;
    return r;
}

extern "C" void kernel_launch(void* const* d_in, const int* in_sizes, int n_in,
                              void* d_out, int out_size, void* d_ws, size_t ws_size,
                              hipStream_t stream) {
    (void)in_sizes; (void)n_in; (void)out_size; (void)ws_size;

    const float* a0   = (const float*)d_in[0];
    const float* sIn  = (const float*)d_in[1];
    const float* z    = (const float*)d_in[2];
    const float* mask = (const float*)d_in[3];
    // params (dict insertion order)
    const float* at_ln_s_g = (const float*)d_in[4];
    const float* at_ada_Wg = (const float*)d_in[5];
    const float* at_ada_bg = (const float*)d_in[6];
    const float* at_ada_Ws = (const float*)d_in[7];
    const float* at_Wq     = (const float*)d_in[8];
    const float* at_bq     = (const float*)d_in[9];
    const float* at_Wk     = (const float*)d_in[10];
    const float* at_Wv     = (const float*)d_in[11];
    const float* at_zln_g  = (const float*)d_in[12];
    const float* at_zln_b  = (const float*)d_in[13];
    const float* at_Wb     = (const float*)d_in[14];
    const float* at_Wgate  = (const float*)d_in[15];
    const float* at_Wo     = (const float*)d_in[16];
    const float* at_Wsg    = (const float*)d_in[17];
    const float* at_bsg    = (const float*)d_in[18];
    const float* tr_ln_s_g = (const float*)d_in[19];
    const float* tr_ada_Wg = (const float*)d_in[20];
    const float* tr_ada_bg = (const float*)d_in[21];
    const float* tr_ada_Ws = (const float*)d_in[22];
    const float* tr_Wa     = (const float*)d_in[23];
    const float* tr_Wb     = (const float*)d_in[24];
    const float* tr_Wout   = (const float*)d_in[25];
    const float* tr_Wsg    = (const float*)d_in[26];
    const float* tr_bsg    = (const float*)d_in[27];

    // ---- workspace layout ----
    char* wp = (char*)d_ws;
    bf16_t* WgA_T   = (bf16_t*)wsalloc(wp, (size_t)L * CC * 2);
    bf16_t* WsA_T   = (bf16_t*)wsalloc(wp, (size_t)L * CC * 2);
    bf16_t* WqT     = (bf16_t*)wsalloc(wp, (size_t)L * CC * 2);
    bf16_t* WkT     = (bf16_t*)wsalloc(wp, (size_t)L * CC * 2);
    bf16_t* WvT     = (bf16_t*)wsalloc(wp, (size_t)L * CC * 2);
    bf16_t* WgateT  = (bf16_t*)wsalloc(wp, (size_t)L * CC * 2);
    bf16_t* WoT     = (bf16_t*)wsalloc(wp, (size_t)L * CC * 2);
    bf16_t* WsgAT   = (bf16_t*)wsalloc(wp, (size_t)L * CC * 2);
    bf16_t* WbpT    = (bf16_t*)wsalloc(wp, (size_t)L * H * CP * 2);
    bf16_t* trWgT   = (bf16_t*)wsalloc(wp, (size_t)L * CC * 2);
    bf16_t* trWsT   = (bf16_t*)wsalloc(wp, (size_t)L * CC * 2);
    bf16_t* trWaT   = (bf16_t*)wsalloc(wp, (size_t)L * CH * 2);
    bf16_t* trWbT   = (bf16_t*)wsalloc(wp, (size_t)L * CH * 2);
    bf16_t* trWoutT = (bf16_t*)wsalloc(wp, (size_t)L * CH * 2);
    bf16_t* trWsgT  = (bf16_t*)wsalloc(wp, (size_t)L * CC * 2);
    float*  zconst  = (float*)wsalloc(wp, (size_t)L * H * 4);
    bf16_t* s_bf    = (bf16_t*)wsalloc(wp, (size_t)NC * 2);
    bf16_t* sln_bf  = (bf16_t*)wsalloc(wp, (size_t)NC * 2);
    float*  gate_at = (float*)wsalloc(wp, (size_t)L * NC * 4);
    float*  at_sig  = (float*)wsalloc(wp, (size_t)L * NC * 4);   // pre-sigmoid
    float*  at_skip = (float*)wsalloc(wp, (size_t)L * NC * 4);
    float*  aA      = (float*)wsalloc(wp, (size_t)NC * 4);
    float*  aln     = (float*)wsalloc(wp, (size_t)NC * 4);
    bf16_t* x_bf    = (bf16_t*)wsalloc(wp, (size_t)NC * 2);
    bf16_t* qbf     = (bf16_t*)wsalloc(wp, (size_t)H * N * DHP * 2);
    bf16_t* kbf     = (bf16_t*)wsalloc(wp, (size_t)H * N * DHP * 2);
    bf16_t* vtbf    = (bf16_t*)wsalloc(wp, (size_t)H * DHP * N * 2);
    float*  gateh   = (float*)wsalloc(wp, (size_t)NC * 4);
    float*  zbuf    = (float*)wsalloc(wp, (size_t)H * N * N * 4);
    bf16_t* og_bf   = (bf16_t*)wsalloc(wp, (size_t)NC * 2);
    float*  b_f32   = (float*)wsalloc(wp, (size_t)NC * 4);
    bf16_t* b_bf    = (bf16_t*)wsalloc(wp, (size_t)NC * 2);
    bf16_t* bln_bf  = (bf16_t*)wsalloc(wp, (size_t)NC * 2);
    float*  S1      = (float*)wsalloc(wp, (size_t)NC * 4);
    float*  S2      = (float*)wsalloc(wp, (size_t)NC * 4);
    bf16_t* xt_bf   = (bf16_t*)wsalloc(wp, (size_t)NC * 2);
    float*  h1      = (float*)wsalloc(wp, (size_t)NHID * 4);
    bf16_t* hid_bf  = (bf16_t*)wsalloc(wp, (size_t)NHID * 2);
    float*  tgate   = (float*)wsalloc(wp, (size_t)NC * 4);

    auto prep = [&](const float* src, const float* rowscale, bf16_t* dst, int K, int Nc) {
        int n = K * Nc;
        prep_w_kernel<<<(n + 255) / 256, 256, 0, stream>>>(src, rowscale, dst, K, Nc);
    };

    // ---- one-time prep ----
    for (int l = 0; l < L; l++) {
        prep(at_ada_Wg + (size_t)l * CC, at_ln_s_g + l * C, WgA_T + (size_t)l * CC, C, C);
        prep(at_ada_Ws + (size_t)l * CC, at_ln_s_g + l * C, WsA_T + (size_t)l * CC, C, C);
        prep(at_Wq + (size_t)l * CC, nullptr, WqT + (size_t)l * CC, C, C);
        prep(at_Wk + (size_t)l * CC, nullptr, WkT + (size_t)l * CC, C, C);
        prep(at_Wv + (size_t)l * CC, nullptr, WvT + (size_t)l * CC, C, C);
        prep(at_Wgate + (size_t)l * CC, nullptr, WgateT + (size_t)l * CC, C, C);
        prep(at_Wo + (size_t)l * CC, nullptr, WoT + (size_t)l * CC, C, C);
        prep(at_Wsg + (size_t)l * CC, nullptr, WsgAT + (size_t)l * CC, C, C);
        prep(at_Wb + (size_t)l * CP * H, at_zln_g + l * CP, WbpT + (size_t)l * H * CP, CP, H);
        prep(tr_ada_Wg + (size_t)l * CC, tr_ln_s_g + l * C, trWgT + (size_t)l * CC, C, C);
        prep(tr_ada_Ws + (size_t)l * CC, tr_ln_s_g + l * C, trWsT + (size_t)l * CC, C, C);
        prep(tr_Wa + (size_t)l * CH, nullptr, trWaT + (size_t)l * CH, C, HID);
        prep(tr_Wb + (size_t)l * CH, nullptr, trWbT + (size_t)l * CH, C, HID);
        prep(tr_Wout + (size_t)l * CH, nullptr, trWoutT + (size_t)l * CH, HID, C);
        prep(tr_Wsg + (size_t)l * CC, nullptr, trWsgT + (size_t)l * CC, C, C);
    }
    zconst_kernel<<<1, 64, 0, stream>>>(at_zln_b, at_Wb, zconst);

    cvt_bf_kernel<<<NC / 256, 256, 0, stream>>>(sIn, s_bf, NC);
    ln_rows_kernel<<<N, 32, 0, stream>>>(sIn, nullptr, sln_bf, C);
    zero_bf_kernel<<<(H * N * DHP) / 256, 256, 0, stream>>>(qbf, H * N * DHP);
    zero_bf_kernel<<<(H * N * DHP) / 256, 256, 0, stream>>>(kbf, H * N * DHP);
    zero_bf_kernel<<<(H * N * DHP) / 256, 256, 0, stream>>>(vtbf, H * N * DHP);

    // s-conditioned precomputes (layer-invariant cond = s)
    for (int l = 0; l < L; l++) {
        launch_gemm(1, s_bf, WsgAT + (size_t)l * CC, at_bsg + l * C, 1.f, N, C, C,
                    gate_at + (size_t)l * NC, nullptr, nullptr, nullptr, stream);
        launch_gemm(0, sln_bf, WgA_T + (size_t)l * CC, at_ada_bg + l * C, 1.f, N, C, C,
                    at_sig + (size_t)l * NC, nullptr, nullptr, nullptr, stream);
        launch_gemm(0, sln_bf, WsA_T + (size_t)l * CC, nullptr, 1.f, N, C, C,
                    at_skip + (size_t)l * NC, nullptr, nullptr, nullptr, stream);
    }

    hipMemcpyAsync(aA, a0, (size_t)NC * sizeof(float), hipMemcpyDeviceToDevice, stream);

    const float qscale = 1.0f / sqrtf((float)DH);

    // ---- layer loop ----
    for (int l = 0; l < L; l++) {
        ln_rows_kernel<<<N, 32, 0, stream>>>(aA, aln, nullptr, C);
        ew_combine_kernel<<<NC / 256, 256, 0, stream>>>(at_sig + (size_t)l * NC, aln,
                                                        at_skip + (size_t)l * NC, x_bf, NC);
        zbias_kernel<<<(N * N) / 256, 256, 0, stream>>>(z, WbpT + (size_t)l * H * CP,
                                                        zconst + l * H, mask, zbuf);
        launch_gemm(3, x_bf, WqT + (size_t)l * CC, at_bq + l * C, qscale, N, C, C,
                    nullptr, qbf, nullptr, nullptr, stream);
        launch_gemm(3, x_bf, WkT + (size_t)l * CC, nullptr, 1.f, N, C, C,
                    nullptr, kbf, nullptr, nullptr, stream);
        launch_gemm(4, x_bf, WvT + (size_t)l * CC, nullptr, 1.f, N, C, C,
                    nullptr, vtbf, nullptr, nullptr, stream);
        launch_gemm(0, x_bf, WgateT + (size_t)l * CC, nullptr, 1.f, N, C, C,
                    gateh, nullptr, nullptr, nullptr, stream);
        attn_kernel<<<(H * (N / 16)) / 4, 128, 0, stream>>>(qbf, kbf, vtbf, zbuf, gateh, og_bf);
        launch_gemm(5, og_bf, WoT + (size_t)l * CC, nullptr, 1.f, N, C, C,
                    b_f32, b_bf, gate_at + (size_t)l * NC, nullptr, stream);
        // transition (cond = b)
        ln_rows_kernel<<<N, 32, 0, stream>>>(b_f32, nullptr, bln_bf, C);
        launch_gemm(0, bln_bf, trWgT + (size_t)l * CC, tr_ada_bg + l * C, 1.f, N, C, C,
                    S1, nullptr, nullptr, nullptr, stream);
        launch_gemm(0, bln_bf, trWsT + (size_t)l * CC, nullptr, 1.f, N, C, C,
                    S2, nullptr, nullptr, nullptr, stream);
        ew_combine_kernel<<<NC / 256, 256, 0, stream>>>(S1, aln, S2, xt_bf, NC);
        launch_gemm(0, xt_bf, trWaT + (size_t)l * CH, nullptr, 1.f, N, C, HID,
                    h1, nullptr, nullptr, nullptr, stream);
        launch_gemm(6, xt_bf, trWbT + (size_t)l * CH, nullptr, 1.f, N, C, HID,
                    nullptr, hid_bf, h1, nullptr, stream);
        launch_gemm(1, b_bf, trWsgT + (size_t)l * CC, tr_bsg + l * C, 1.f, N, C, C,
                    tgate, nullptr, nullptr, nullptr, stream);
        float* aout = (l == L - 1) ? (float*)d_out : aA;
        launch_gemm(7, hid_bf, trWoutT + (size_t)l * CH, nullptr, 1.f, N, HID, C,
                    aout, nullptr, b_f32, tgate, stream);
    }
}
